// mhsa_rope_37220186587431
// MI455X (gfx1250) — matrix operand order repaired and checked by reading, not compiled, not run on hardware
//
#include <hip/hip_runtime.h>
#include <hip/hip_bf16.h>

typedef __attribute__((ext_vector_type(16))) __bf16 v16bf;
typedef __attribute__((ext_vector_type(8)))  __bf16 v8bf;
typedef __attribute__((ext_vector_type(8)))  float  v8f;
typedef __attribute__((ext_vector_type(4)))  unsigned int v4u;
typedef __attribute__((ext_vector_type(8)))  int v8i;
typedef __attribute__((ext_vector_type(4)))  int v4i;

#define D_MODEL 2048
#define NUM_HEADS 16
#define D_K 128
#define SEQ 2048
#define BATCH 2
#define MROWS (BATCH * SEQ)   // 4096

#define WMMA_BF16(a, b, c) \
  __builtin_amdgcn_wmma_f32_16x16x32_bf16(false, (a), false, (b), (short)0, (c), false, false)

__device__ __forceinline__ unsigned short f32_to_bf16_rne(float f) {
  unsigned u = __float_as_uint(f);
  unsigned r = 0x7FFFu + ((u >> 16) & 1u);
  return (unsigned short)((u + r) >> 16);
}
__device__ __forceinline__ float bf16_to_f32(unsigned short h) {
  return __uint_as_float(((unsigned)h) << 16);
}

// A fragment (16 rows x 32 K, bf16). Contiguous k order used here: lane L holds row L&15;
// low half-wave K=0..15, high half-wave K=16..31.
__device__ __forceinline__ v16bf load_a_frag(const unsigned short* base, int stride, int lane) {
  int row = lane & 15;
  int kb  = (lane >> 4) << 4;   // 0 or 16
  const unsigned short* p = base + (size_t)row * stride + kb;
  union { v16bf v; v8bf h[2]; } u;
  u.h[0] = *(const v8bf*)(p);
  u.h[1] = *(const v8bf*)(p + 8);
  return u.v;
}

// B fragment (32 K x 16 cols, bf16), supplied as row-major Bt[col][k].
// Contiguous k order used here: lane L holds col L&15; low half-wave K=0..15, high half-wave K=16..31.
__device__ __forceinline__ v16bf load_b_frag(const unsigned short* base, int stride, int lane) {
  int col = lane & 15;
  int kb  = (lane >> 4) << 4;   // 0 or 16
  return *(const v16bf*)(base + (size_t)col * stride + kb);
}

// ------------------------------------------------------------ TDM 2-D tile load
// Builds the Tensor DMA Descriptor (D#) per CDNA5 ISA §8 and issues
// tensor_load_to_lds. 2-byte elements, 2-D tile (groups 2/3 zero).
__device__ __forceinline__ void tdm_load_2d(unsigned lds_off, const void* gptr,
                                            unsigned tile_d0, unsigned tile_d1,
                                            unsigned tensor_d0, unsigned tensor_d1,
                                            unsigned long long stride0) {
  unsigned long long ga = (unsigned long long)gptr;
  v4u g0;
  g0[0] = 1u;                                              // count=1, user descriptor
  g0[1] = lds_off;                                         // lds_addr (bytes)
  g0[2] = (unsigned)(ga & 0xffffffffu);                    // global_addr[31:0]
  g0[3] = (unsigned)((ga >> 32) & 0x01ffffffu) | 0x80000000u; // addr[56:32] | type=2
  v8i g1;
  g1[0] = (int)(1u << 16);                                 // data_size = 2 bytes
  g1[1] = (int)((tensor_d0 & 0xffffu) << 16);              // tensor_dim0[15:0]
  g1[2] = (int)(((tensor_d0 >> 16) & 0xffffu) | ((tensor_d1 & 0xffffu) << 16));
  g1[3] = (int)(((tensor_d1 >> 16) & 0xffffu) | ((tile_d0 & 0xffffu) << 16));
  g1[4] = (int)(tile_d1 & 0xffffu);                        // tile_dim1 (tile_dim2=0)
  g1[5] = (int)(stride0 & 0xffffffffull);                  // tensor_dim0_stride lo
  g1[6] = (int)((stride0 >> 32) & 0xffffull);              // stride hi (dim1_stride=0)
  g1[7] = 0;
  v4i z = {0, 0, 0, 0};
#if __clang_major__ >= 23
  v8i z8 = {0, 0, 0, 0, 0, 0, 0, 0};
  __builtin_amdgcn_tensor_load_to_lds(g0, g1, z, z, z8, 0);
#else
  __builtin_amdgcn_tensor_load_to_lds(g0, g1, z, z, 0);
#endif
}

// ---------------------------------------------------------------- conversion
__global__ void cvt_f32_bf16(const float* __restrict__ in, unsigned short* __restrict__ out, int n4) {
  int i = blockIdx.x * blockDim.x + threadIdx.x;
  if (i < n4) {
    float4 v = ((const float4*)in)[i];
    ushort4 o;
    o.x = f32_to_bf16_rne(v.x); o.y = f32_to_bf16_rne(v.y);
    o.z = f32_to_bf16_rne(v.z); o.w = f32_to_bf16_rne(v.w);
    ((ushort4*)out)[i] = o;
  }
}

// ---------------------------------------------------------------- GEMM
// C[M][N] = A[M][K] * W[N][K]^T (bf16, K-major both sides). Block 128x128,
// 8 waves as 4(M) x 2(N); each wave 32x64 = 2x4 WMMA accumulators.
// MODE 0: f32 row-major out. MODE 1: bf16 -> [b,h,s,d]. MODE 2: bf16 -> [b,h,d,s].
template <int MODE>
__global__ __launch_bounds__(256)
void gemm_bf16_kernel(const unsigned short* __restrict__ A, const unsigned short* __restrict__ W,
                      float* __restrict__ outF, unsigned short* __restrict__ outB, int Kdim) {
  const int lane = threadIdx.x & 31;
  const int wave = threadIdx.x >> 5;
  const int wm = wave >> 1;
  const int wn = wave & 1;
  const int m0 = blockIdx.y * 128 + wm * 32;
  const int n0 = blockIdx.x * 128 + wn * 64;

  v8f acc[2][4];
#pragma unroll
  for (int i = 0; i < 2; ++i)
#pragma unroll
    for (int j = 0; j < 4; ++j) acc[i][j] = (v8f){0.f, 0.f, 0.f, 0.f, 0.f, 0.f, 0.f, 0.f};

  for (int k0 = 0; k0 < Kdim; k0 += 32) {
    v16bf af[2], bfr[4];
#pragma unroll
    for (int mt = 0; mt < 2; ++mt)
      af[mt] = load_a_frag(A + (size_t)(m0 + mt * 16) * Kdim + k0, Kdim, lane);
#pragma unroll
    for (int nt = 0; nt < 4; ++nt)
      bfr[nt] = load_b_frag(W + (size_t)(n0 + nt * 16) * Kdim + k0, Kdim, lane);
#pragma unroll
    for (int mt = 0; mt < 2; ++mt)
#pragma unroll
      for (int nt = 0; nt < 4; ++nt)
        acc[mt][nt] = WMMA_BF16(af[mt], bfr[nt], acc[mt][nt]);
  }

  const int ncol = lane & 15;
  const int mhi  = (lane >> 4) << 3;   // C layout: M = r + 8*(lane>=16), N = lane&15
#pragma unroll
  for (int mt = 0; mt < 2; ++mt) {
#pragma unroll
    for (int nt = 0; nt < 4; ++nt) {
#pragma unroll
      for (int r = 0; r < 8; ++r) {
        int m = m0 + mt * 16 + mhi + r;
        int n = n0 + nt * 16 + ncol;
        float v = acc[mt][nt][r];
        if (MODE == 0) {
          outF[(size_t)m * D_MODEL + n] = v;
        } else {
          int b = m >> 11, s = m & (SEQ - 1);
          int h = n >> 7,  d = n & (D_K - 1);
          unsigned short bv = f32_to_bf16_rne(v);
          if (MODE == 1)
            outB[((size_t)(b * NUM_HEADS + h) * SEQ + s) * D_K + d] = bv;   // [b,h,s,d]
          else
            outB[((size_t)(b * NUM_HEADS + h) * D_K + d) * SEQ + s] = bv;   // [b,h,d,s]
        }
      }
    }
  }
}

// ---------------------------------------------------------------- RoPE on Q and K ([b,h,s,d])
__global__ void rope_kernel(unsigned short* __restrict__ Q, unsigned short* __restrict__ K,
                            const int* __restrict__ pos) {
  int idx = blockIdx.x * blockDim.x + threadIdx.x;
  if (idx >= BATCH * NUM_HEADS * SEQ * (D_K / 2)) return;
  int i  = idx & 63;
  int s  = (idx >> 6) & (SEQ - 1);
  int bh = idx >> 17;
  float freq = __expf(-(float)i * (9.2103403719761836f / 64.0f));  // theta^(-2i/128)
  float ang = (float)pos[s] * freq;
  float sn, cs;
  __sincosf(ang, &sn, &cs);
  size_t base = ((size_t)bh * SEQ + s) * D_K + 2 * i;

  float q1 = bf16_to_f32(Q[base]), q2 = bf16_to_f32(Q[base + 1]);
  Q[base]     = f32_to_bf16_rne(q1 * cs - q2 * sn);
  Q[base + 1] = f32_to_bf16_rne(q1 * sn + q2 * cs);
  float k1 = bf16_to_f32(K[base]), k2 = bf16_to_f32(K[base + 1]);
  K[base]     = f32_to_bf16_rne(k1 * cs - k2 * sn);
  K[base + 1] = f32_to_bf16_rne(k1 * sn + k2 * cs);
}

// ---------------------------------------------------------------- flash attention
// Grid: (S/128 q-tiles, B*H), 256 threads (8 waves). Block-cooperative kv loop
// with DOUBLE-BUFFERED Tensor Data Mover staging: wave 0 issues the K (64x128)
// and V^T (128x64) tile DMAs for tile t+1, then waits TENSORcnt<=2 (in-order
// per-wave completion => tile t landed), barrier, all 8 waves compute tile t.
// Dynamic LDS: [K0|K1|V0|V1|P] = 5 x 16 KB = 80 KB (4 blocks still fit a WGP).
// Q,K: [b,h,s,d] bf16.  Vt: [b,h,d,s] bf16.  O: [b,s,d_model] bf16.
#define KBUF 8192   // 16 KB / sizeof(ushort)

__global__ __launch_bounds__(256)
void flash_attn_kernel(const unsigned short* __restrict__ Q, const unsigned short* __restrict__ K,
                       const unsigned short* __restrict__ Vt, unsigned short* __restrict__ O) {
  extern __shared__ unsigned short lds[];
  const int lane = threadIdx.x & 31;
  const int wave = threadIdx.x >> 5;
  const int bh = blockIdx.y;
  const int q0b = blockIdx.x * 128;
  const int q0w = q0b + wave * 16;

  const unsigned short* Qb = Q  + ((size_t)bh * SEQ + q0w) * D_K;
  const unsigned short* Kb = K  + (size_t)bh * SEQ * D_K;
  const unsigned short* Vb = Vt + (size_t)bh * D_K * SEQ;
  unsigned short* myp = lds + 4 * KBUF + wave * (16 * 64);
  const unsigned ldsBase = (unsigned)(unsigned long long)(void*)lds;

  v16bf qf[4];
#pragma unroll
  for (int ks = 0; ks < 4; ++ks) qf[ks] = load_a_frag(Qb + ks * 32, D_K, lane);

  v8f acc_o[8];
#pragma unroll
  for (int dt = 0; dt < 8; ++dt) acc_o[dt] = (v8f){0.f, 0.f, 0.f, 0.f, 0.f, 0.f, 0.f, 0.f};
  float mrow[8], lrow[8];
#pragma unroll
  for (int r = 0; r < 8; ++r) { mrow[r] = -1e30f; lrow[r] = 0.f; }

  const float scale = 0.08838834764831845f;  // 1/sqrt(128)
  const int ncol = lane & 15;
  const int mhi  = (lane >> 4) << 3;
  const int kv_tiles = (q0b + 128) >> 6;     // block-uniform causal bound

  // prologue: stage tile 0 into buffer 0
  if (wave == 0) {
    tdm_load_2d(ldsBase, Kb, D_K, 64, D_K, SEQ, D_K);
    tdm_load_2d(ldsBase + 2u * 2 * KBUF, Vb, 64, D_K, SEQ, D_K, SEQ);
  }

  for (int t = 0; t < kv_tiles; ++t) {
    const int kv0 = t << 6;
    if (wave == 0) {
      if (t + 1 < kv_tiles) {
        const int kv1 = kv0 + 64;
        const unsigned buf = (unsigned)((t + 1) & 1);
        tdm_load_2d(ldsBase + buf * 2 * KBUF, Kb + (size_t)kv1 * D_K,
                    D_K, 64, D_K, (unsigned)(SEQ - kv1), D_K);
        tdm_load_2d(ldsBase + (2u + buf) * 2 * KBUF, Vb + kv1,
                    64, D_K, (unsigned)(SEQ - kv1), D_K, SEQ);
        __builtin_amdgcn_s_wait_tensorcnt(2);   // tile t's two DMAs done
      } else {
        __builtin_amdgcn_s_wait_tensorcnt(0);
      }
    }
    __syncthreads();                            // buf[t&1] visible to all waves

    const unsigned short* ldsK = lds + (t & 1) * KBUF;           // [kv][d]
    const unsigned short* ldsV = lds + (2 + (t & 1)) * KBUF;     // [d][kv]

    if (kv0 < q0w + 16) {                       // tile relevant for this wave (causal)
      v8f sacc[4];
#pragma unroll
      for (int nt = 0; nt < 4; ++nt) sacc[nt] = (v8f){0.f, 0.f, 0.f, 0.f, 0.f, 0.f, 0.f, 0.f};

      // S = Q * K^T (16 x 64 per wave), K from LDS
#pragma unroll
      for (int nt = 0; nt < 4; ++nt) {
#pragma unroll
        for (int ks = 0; ks < 4; ++ks) {
          v16bf kf = load_b_frag(ldsK + (nt * 16) * D_K + ks * 32, D_K, lane);
          sacc[nt] = WMMA_BF16(qf[ks], kf, sacc[nt]);
        }
      }

      // scale + causal mask
      const bool need_mask = (kv0 + 64) > q0w;
#pragma unroll
      for (int nt = 0; nt < 4; ++nt) {
        int col = kv0 + nt * 16 + ncol;
#pragma unroll
        for (int r = 0; r < 8; ++r) {
          float s = sacc[nt][r] * scale;
          if (need_mask && col > (q0w + mhi + r)) s = -1e30f;
          sacc[nt][r] = s;
        }
      }

      // online softmax (16-lane xor reductions stay inside half-wave)
      float mnew[8], corr[8];
#pragma unroll
      for (int r = 0; r < 8; ++r) {
        float v = fmaxf(fmaxf(sacc[0][r], sacc[1][r]), fmaxf(sacc[2][r], sacc[3][r]));
        v = fmaxf(v, __shfl_xor(v, 1, 32));
        v = fmaxf(v, __shfl_xor(v, 2, 32));
        v = fmaxf(v, __shfl_xor(v, 4, 32));
        v = fmaxf(v, __shfl_xor(v, 8, 32));
        mnew[r] = fmaxf(mrow[r], v);
      }

      float rsum[8];
#pragma unroll
      for (int r = 0; r < 8; ++r) rsum[r] = 0.f;
#pragma unroll
      for (int nt = 0; nt < 4; ++nt) {
#pragma unroll
        for (int r = 0; r < 8; ++r) {
          float p = __expf(sacc[nt][r] - mnew[r]);
          rsum[r] += p;
          myp[(mhi + r) * 64 + nt * 16 + ncol] = f32_to_bf16_rne(p);
        }
      }
#pragma unroll
      for (int r = 0; r < 8; ++r) {
        float v = rsum[r];
        v += __shfl_xor(v, 1, 32);
        v += __shfl_xor(v, 2, 32);
        v += __shfl_xor(v, 4, 32);
        v += __shfl_xor(v, 8, 32);
        float c = __expf(mrow[r] - mnew[r]);
        lrow[r] = lrow[r] * c + v;
        mrow[r] = mnew[r];
        corr[r] = c;
      }
#pragma unroll
      for (int dt = 0; dt < 8; ++dt)
#pragma unroll
        for (int r = 0; r < 8; ++r) acc_o[dt][r] *= corr[r];

      // O += P * V : P from per-wave LDS (A frags), V from shared LDS tile (B frags)
#pragma unroll
      for (int ks2 = 0; ks2 < 2; ++ks2) {
        v16bf af = load_a_frag(myp + ks2 * 32, 64, lane);
#pragma unroll
        for (int dt = 0; dt < 8; ++dt) {
          v16bf vf = load_b_frag(ldsV + (dt * 16) * 64 + ks2 * 32, 64, lane);
          acc_o[dt] = WMMA_BF16(af, vf, acc_o[dt]);
        }
      }
    }
    __syncthreads();                            // buf[t&1] free before overwrite
  }

  // epilogue: O[b, s, h*128 + d]
  const int b = bh >> 4, h = bh & 15;
#pragma unroll
  for (int dt = 0; dt < 8; ++dt) {
#pragma unroll
    for (int r = 0; r < 8; ++r) {
      int srow = q0w + mhi + r;
      float o = acc_o[dt][r] / lrow[r];
      O[((size_t)(b * SEQ + srow)) * D_MODEL + h * D_K + dt * 16 + ncol] = f32_to_bf16_rne(o);
    }
  }
}

// ---------------------------------------------------------------- launch
extern "C" void kernel_launch(void* const* d_in, const int* in_sizes, int n_in,
                              void* d_out, int out_size, void* d_ws, size_t ws_size,
                              hipStream_t stream) {
  const float* x  = (const float*)d_in[0];
  const int* pos  = (const int*)d_in[1];
  const float* wq = (const float*)d_in[2];
  const float* wk = (const float*)d_in[3];
  const float* wv = (const float*)d_in[4];
  const float* wo = (const float*)d_in[5];
  float* out = (float*)d_out;

  const size_t SZ_X = (size_t)MROWS * D_MODEL * 2;       // 16 MB
  const size_t SZ_W = (size_t)D_MODEL * D_MODEL * 2;     // 8 MB
  char* ws = (char*)d_ws;
  unsigned short* Xb  = (unsigned short*)(ws);
  unsigned short* Wqb = (unsigned short*)(ws + SZ_X);
  unsigned short* Wkb = (unsigned short*)(ws + SZ_X + SZ_W);
  unsigned short* Wvb = (unsigned short*)(ws + SZ_X + 2 * SZ_W);
  unsigned short* Wob = (unsigned short*)(ws + SZ_X + 3 * SZ_W);
  unsigned short* Qb  = (unsigned short*)(ws + SZ_X + 4 * SZ_W);
  unsigned short* Kbf = (unsigned short*)(ws + 2 * SZ_X + 4 * SZ_W);
  unsigned short* Vtb = (unsigned short*)(ws + 3 * SZ_X + 4 * SZ_W);
  unsigned short* Ob  = (unsigned short*)(ws + 4 * SZ_X + 4 * SZ_W);
  if (ws_size < 5 * SZ_X + 4 * SZ_W) return;

  const int nX4 = MROWS * D_MODEL / 4;
  const int nW4 = D_MODEL * D_MODEL / 4;
  cvt_f32_bf16<<<(nX4 + 255) / 256, 256, 0, stream>>>(x,  Xb,  nX4);
  cvt_f32_bf16<<<(nW4 + 255) / 256, 256, 0, stream>>>(wq, Wqb, nW4);
  cvt_f32_bf16<<<(nW4 + 255) / 256, 256, 0, stream>>>(wk, Wkb, nW4);
  cvt_f32_bf16<<<(nW4 + 255) / 256, 256, 0, stream>>>(wv, Wvb, nW4);
  cvt_f32_bf16<<<(nW4 + 255) / 256, 256, 0, stream>>>(wo, Wob, nW4);

  dim3 ggrid(D_MODEL / 128, MROWS / 128);   // (16, 32)
  gemm_bf16_kernel<1><<<ggrid, 256, 0, stream>>>(Xb, Wqb, nullptr, Qb,  D_MODEL);
  gemm_bf16_kernel<1><<<ggrid, 256, 0, stream>>>(Xb, Wkb, nullptr, Kbf, D_MODEL);
  gemm_bf16_kernel<2><<<ggrid, 256, 0, stream>>>(Xb, Wvb, nullptr, Vtb, D_MODEL);

  const int nR = BATCH * NUM_HEADS * SEQ * (D_K / 2);
  rope_kernel<<<(nR + 255) / 256, 256, 0, stream>>>(Qb, Kbf, pos);

  // 80 KB dynamic LDS for double-buffered TDM staging + P tiles
  const int flash_lds = 5 * KBUF * (int)sizeof(unsigned short);
  (void)hipFuncSetAttribute((const void*)flash_attn_kernel,
                            hipFuncAttributeMaxDynamicSharedMemorySize, flash_lds);
  dim3 fgrid(SEQ / 128, BATCH * NUM_HEADS);              // (16, 32)
  flash_attn_kernel<<<fgrid, 256, flash_lds, stream>>>(Qb, Kbf, Vtb, Ob);

  gemm_bf16_kernel<0><<<ggrid, 256, 0, stream>>>(Ob, Wob, out, nullptr, D_MODEL);
}
